// SMPLlayer_24988119728633
// MI455X (gfx1250) — matrix operand order repaired and checked by reading, not compiled, not run on hardware
//
#include <hip/hip_runtime.h>

typedef __attribute__((ext_vector_type(16))) __bf16 v16bf;
typedef __attribute__((ext_vector_type(8)))  __bf16 v8bf;
typedef __attribute__((ext_vector_type(8)))  float  v8f;

#define B_   512
#define V_   6890
#define NJ_  24
#define S_   10
#define N3   20670          // V_*3
#define KX   224            // 10 (shapes) + 207 (pose feature) padded to 7*32
#define VPAD 6896           // 431 tiles * 16
#define NPAD 20736          // 81 gemm chunks * 256 = 162 pack chunks * 128
#define NCHP 162            // pack chunks (128 rows)
#define NCHG 81             // gemm chunks (256 cols)

// workspace byte offsets (all 32B aligned)
#define OFF_JS   0u          // 24*33 f32
#define OFF_X    4096u       // 512*224 bf16
#define OFF_RELT 233472u     // 512 * (16 x 32) bf16
#define OFF_WGT  757760u     // 6896*32 bf16
#define OFF_WT   1199104u    // 20736*224 bf16 (fused [shapedirs;posedirs], N-major)

__device__ const int kPar[24] = {-1,0,0,0,1,2,3,4,5,6,7,8,9,9,9,12,13,14,16,17,18,19,20,21};

static __device__ inline v16bf cat16(v8bf lo, v8bf hi) {
  return __builtin_shufflevector(lo, hi, 0,1,2,3,4,5,6,7,8,9,10,11,12,13,14,15);
}

// ---------------------------------------------------------------------------
// 1) JS[j][c][l] = sum_v JR[j,v]*shapedirs[v,c,l];  JT[j][c] = sum_v JR[j,v]*vt[v,c]
// ---------------------------------------------------------------------------
__global__ void __launch_bounds__(256) k_joint_reg(
    const float* __restrict__ JR, const float* __restrict__ vt,
    const float* __restrict__ sd, float* __restrict__ js) {
  __shared__ float red[256][34];
  const int j = blockIdx.x, t = threadIdx.x;
  float acc[33];
#pragma unroll
  for (int q = 0; q < 33; ++q) acc[q] = 0.f;
  for (int v = t; v < V_; v += 256) {
    const float r = JR[(size_t)j * V_ + v];
    const float* s = sd + (size_t)v * 30;
#pragma unroll
    for (int q = 0; q < 30; ++q) acc[q] += r * s[q];
    const float* tv = vt + (size_t)v * 3;
    acc[30] += r * tv[0]; acc[31] += r * tv[1]; acc[32] += r * tv[2];
  }
#pragma unroll
  for (int q = 0; q < 33; ++q) red[t][q] = acc[q];
  __syncthreads();
  if (t < 33) {
    float s = 0.f;
    for (int i = 0; i < 256; ++i) s += red[i][t];
    js[j * 33 + t] = s;
  }
}

// ---------------------------------------------------------------------------
// 2) repack skinning weights to bf16, (VPAD x 32), zero padded
// ---------------------------------------------------------------------------
__global__ void __launch_bounds__(256) k_wpack(
    const float* __restrict__ w, __bf16* __restrict__ wgt) {
  const int g = blockIdx.x * 256 + threadIdx.x;
  if (g >= VPAD * 32) return;
  const int v = g >> 5, jj = g & 31;
  const float val = (v < V_ && jj < NJ_) ? w[(size_t)v * NJ_ + jj] : 0.f;
  wgt[g] = (__bf16)val;
}

// ---------------------------------------------------------------------------
// 3) one-time pack of fused GEMM weights into bf16 N-major Wt[NPAD][224]
// ---------------------------------------------------------------------------
__global__ void __launch_bounds__(256) k_wtpack(
    const float* __restrict__ sd, const float* __restrict__ pd,
    __bf16* __restrict__ wt) {
  __shared__ __bf16 T[128 * 232];            // 128 rows, stride 232 bf16 (116 dw)
  const int tid = threadIdx.x;
  const int nbase = blockIdx.x * 128;
  for (int i = tid; i < 128 * 224; i += 256) {
    const int nl = i & 127, kg = i >> 7;
    const int n = nbase + nl;
    float v = 0.f;
    if (n < N3) {
      if (kg < 10)        v = sd[(size_t)n * 10 + kg];
      else if (kg < 217) {
        v = pd[(size_t)(kg - 10) * N3 + n];
        if (kg + 12 < 217) __builtin_prefetch(&pd[(size_t)(kg + 2) * N3 + n], 0, 1);
      }
    }
    T[nl * 232 + kg] = (__bf16)v;
  }
  __syncthreads();
  unsigned int* wtd = (unsigned int*)wt;
  const unsigned int* Td = (const unsigned int*)T;
  for (int i = tid; i < 128 * 112; i += 256) {
    const int nl = i / 112, c2 = i % 112;
    wtd[(size_t)(nbase + nl) * 112 + c2] = Td[nl * 116 + c2];
  }
}

// ---------------------------------------------------------------------------
// 4) per-batch: rodrigues, pose-feature row of X, kinematic chain,
//    posed_joints output, relT (bf16, 16x32 per batch) for the skinning WMMA.
// ---------------------------------------------------------------------------
__global__ void __launch_bounds__(256) k_pose(
    const float* __restrict__ poses, const float* __restrict__ shapes,
    const float* __restrict__ js, __bf16* __restrict__ X,
    __bf16* __restrict__ relT, float* __restrict__ out) {
  __shared__ float sTm[8][24][12];
  __shared__ float sA [8][24][12];
  __shared__ float sJ [8][24][3];
  const int w = threadIdx.x >> 5, lane = threadIdx.x & 31;
  const int b = blockIdx.x * 8 + w;

  float sh[10];
#pragma unroll
  for (int l = 0; l < 10; ++l) sh[l] = shapes[(size_t)b * S_ + l];

  float R[9];
  if (lane < NJ_) {
    const float rx = poses[(size_t)b * 72 + lane * 3 + 0];
    const float ry = poses[(size_t)b * 72 + lane * 3 + 1];
    const float rz = poses[(size_t)b * 72 + lane * 3 + 2];
    const float ang = sqrtf(rx * rx + ry * ry + rz * rz + 1e-8f);
    const float inv = 1.f / ang;
    const float ax = rx * inv, ay = ry * inv, az = rz * inv;
    const float cn = cosf(ang), sn = sinf(ang), oc = 1.f - cn;
    R[0] = cn + oc * ax * ax;  R[1] = oc * ax * ay - sn * az;  R[2] = oc * ax * az + sn * ay;
    R[3] = oc * ay * ax + sn * az;  R[4] = cn + oc * ay * ay;  R[5] = oc * ay * az - sn * ax;
    R[6] = oc * az * ax - sn * ay;  R[7] = oc * az * ay + sn * ax;  R[8] = cn + oc * az * az;
    const float* jrow = js + lane * 33;
#pragma unroll
    for (int c = 0; c < 3; ++c) {
      float s = jrow[30 + c];
#pragma unroll
      for (int l = 0; l < 10; ++l) s += sh[l] * jrow[c * 10 + l];
      sJ[w][lane][c] = s;
    }
    if (lane >= 1) {  // pose feature -> X cols 10..216
      const size_t base = (size_t)b * KX + 10 + (size_t)(lane - 1) * 9;
#pragma unroll
      for (int e = 0; e < 9; ++e) {
        const float ident = (e == 0 || e == 4 || e == 8) ? 1.f : 0.f;
        X[base + e] = (__bf16)(R[e] - ident);
      }
    }
  }
  if (lane < 10) X[(size_t)b * KX + lane] = (__bf16)sh[lane];
  if (lane >= 24 && lane < 31) X[(size_t)b * KX + 217 + (lane - 24)] = (__bf16)0.f;
  __syncthreads();

  if (lane < NJ_) {
    const int p = kPar[lane];
#pragma unroll
    for (int r = 0; r < 3; ++r) {
      sTm[w][lane][r * 4 + 0] = R[r * 3 + 0];
      sTm[w][lane][r * 4 + 1] = R[r * 3 + 1];
      sTm[w][lane][r * 4 + 2] = R[r * 3 + 2];
      float rj = sJ[w][lane][r];
      if (lane > 0) rj -= sJ[w][p][r];
      sTm[w][lane][r * 4 + 3] = rj;
    }
  }
  __syncthreads();
  if (lane < 12) sA[w][0][lane] = sTm[w][0][lane];
  __syncthreads();
  for (int i = 1; i < NJ_; ++i) {
    if (lane < 12) {
      const int p = kPar[i];
      const int r = lane >> 2, c = lane & 3;
      float v = sA[w][p][r * 4 + 0] * sTm[w][i][0 * 4 + c]
              + sA[w][p][r * 4 + 1] * sTm[w][i][1 * 4 + c]
              + sA[w][p][r * 4 + 2] * sTm[w][i][2 * 4 + c];
      if (c == 3) v += sA[w][p][r * 4 + 3];
      sA[w][i][lane] = v;
    }
    __syncthreads();
  }

  if (lane < NJ_) {  // posed_joints = A[..., :3, 3]
    float* jo = out + (size_t)B_ * N3 + (size_t)(b * NJ_ + lane) * 3;
    jo[0] = sA[w][lane][3]; jo[1] = sA[w][lane][7]; jo[2] = sA[w][lane][11];
  }

  { // relT[b][mn][j] (16 x 32, zero padded)
    __bf16* rb = relT + (size_t)b * 512;
#pragma unroll
    for (int mn = 0; mn < 16; ++mn) {
      float v = 0.f;
      if (mn < 12 && lane < NJ_) {
        const int m = mn >> 2, nn = mn & 3;
        if (nn < 3) v = sA[w][lane][m * 4 + nn];
        else v = sA[w][lane][m * 4 + 3]
               - (sA[w][lane][m * 4 + 0] * sJ[w][lane][0]
                + sA[w][lane][m * 4 + 1] * sJ[w][lane][1]
                + sA[w][lane][m * 4 + 2] * sJ[w][lane][2]);
      }
      rb[mn * 32 + lane] = (__bf16)v;
    }
  }
}

// ---------------------------------------------------------------------------
// 5) v_posed = v_template + [shapes|pf] @ Wt^T :  M=512, N=20670, K=224
//    Block tile 32(M) x 256(N); each wave: 2 A-frags x 2 B-frags -> 4 WMMAs
//    per K-step. b128 staging, LDS ping-pong, explicit load/compute/store split.
// ---------------------------------------------------------------------------
__global__ void __launch_bounds__(256) k_gemm(
    const __bf16* __restrict__ X, const __bf16* __restrict__ wt,
    const float* __restrict__ vtp, float* __restrict__ out) {
  __shared__ __bf16 Xt[32 * KX];            // 14336 B
  __shared__ __bf16 BT[2][256 * 40];        // 2 x 20480 B (row stride 40 bf16 = 5 uint4)
  const int tid = threadIdx.x;
  const int wav = tid >> 5, lane = tid & 31, lrow = lane & 15, hi = lane >> 4;
  const int mtile = blockIdx.y * 32;
  const int nbase = blockIdx.x * 256;
  const uint4* wt4 = (const uint4*)wt;      // row stride 28 uint4

  { // stage X tile (b64 copies)
    const uint2* src = (const uint2*)X;
    uint2* dst = (uint2*)Xt;
#pragma unroll
    for (int i = 0; i < 7; ++i) {
      const int idx = i * 256 + tid;
      const int row = idx / 56, c2 = idx % 56;
      dst[idx] = src[(size_t)(mtile + row) * 56 + c2];
    }
  }
  const int snl = tid >> 2, sc4 = tid & 3;  // staging: thread -> (row, uint4) pairs
  { // stage B slab ks=0 (b128 copies: 256 rows x 4 uint4)
    uint4* dst = (uint4*)BT[0];
#pragma unroll
    for (int i = 0; i < 4; ++i) {
      const int nl = snl + i * 64;
      dst[nl * 5 + sc4] = wt4[(size_t)(nbase + nl) * 28 + sc4];
    }
  }
  __syncthreads();

  const int ng0 = nbase + wav * 32 + lrow;
  const int ng1 = ng0 + 16;
  const float bias0 = (ng0 < N3) ? vtp[ng0] : 0.f;
  const float bias1 = (ng1 < N3) ? vtp[ng1] : 0.f;
  v8f acc00 = {0.f,0.f,0.f,0.f,0.f,0.f,0.f,0.f};
  v8f acc01 = acc00, acc10 = acc00, acc11 = acc00;

  for (int ks = 0; ks < 7; ++ks) {
    // preload next slab into registers (overlaps WMMAs)
    uint4 t0, t1, t2, t3;
    if (ks < 6) {
      const size_t sb = (size_t)(ks + 1) * 4 + sc4;
      t0 = wt4[(size_t)(nbase + snl +   0) * 28 + sb];
      t1 = wt4[(size_t)(nbase + snl +  64) * 28 + sb];
      t2 = wt4[(size_t)(nbase + snl + 128) * 28 + sb];
      t3 = wt4[(size_t)(nbase + snl + 192) * 28 + sb];
    }
    // fragments from current buffer
    const __bf16* ap0 = Xt + lrow * KX + ks * 32;
    const __bf16* ap1 = ap0 + 16 * KX;
    const v16bf a0 = cat16(*(const v8bf*)(ap0 + (hi ? 16 : 0)),
                           *(const v8bf*)(ap0 + (hi ? 24 : 8)));
    const v16bf a1 = cat16(*(const v8bf*)(ap1 + (hi ? 16 : 0)),
                           *(const v8bf*)(ap1 + (hi ? 24 : 8)));
    const __bf16* bp0 = BT[ks & 1] + (wav * 32 + lrow) * 40 + hi * 16;
    const __bf16* bp1 = bp0 + 16 * 40;
    const v16bf b0 = cat16(*(const v8bf*)bp0, *(const v8bf*)(bp0 + 8));
    const v16bf b1 = cat16(*(const v8bf*)bp1, *(const v8bf*)(bp1 + 8));
    acc00 = __builtin_amdgcn_wmma_f32_16x16x32_bf16(false, a0, false, b0, (short)0, acc00, false, false);
    acc10 = __builtin_amdgcn_wmma_f32_16x16x32_bf16(false, a1, false, b0, (short)0, acc10, false, false);
    acc01 = __builtin_amdgcn_wmma_f32_16x16x32_bf16(false, a0, false, b1, (short)0, acc01, false, false);
    acc11 = __builtin_amdgcn_wmma_f32_16x16x32_bf16(false, a1, false, b1, (short)0, acc11, false, false);
    // commit next slab
    if (ks < 6) {
      uint4* dst = (uint4*)BT[(ks + 1) & 1];
      dst[(snl +   0) * 5 + sc4] = t0;
      dst[(snl +  64) * 5 + sc4] = t1;
      dst[(snl + 128) * 5 + sc4] = t2;
      dst[(snl + 192) * 5 + sc4] = t3;
    }
    __syncthreads();
  }
  if (ng0 < N3) {
#pragma unroll
    for (int r = 0; r < 8; ++r) {
      out[(size_t)(mtile +      r + hi * 8) * N3 + ng0] = acc00[r] + bias0;
      out[(size_t)(mtile + 16 + r + hi * 8) * N3 + ng0] = acc10[r] + bias0;
    }
  }
  if (ng1 < N3) {
#pragma unroll
    for (int r = 0; r < 8; ++r) {
      out[(size_t)(mtile +      r + hi * 8) * N3 + ng1] = acc01[r] + bias1;
      out[(size_t)(mtile + 16 + r + hi * 8) * N3 + ng1] = acc11[r] + bias1;
    }
  }
}

// ---------------------------------------------------------------------------
// 6) skinning, in place on d_out. D[mn,v] = sum_j relT[b][mn][j]*wgt[v][j].
//    A-fragment (per batch) is loop-invariant: 4 vertex tiles per wave.
// ---------------------------------------------------------------------------
__global__ void __launch_bounds__(256) k_skin(
    const __bf16* __restrict__ relT, const __bf16* __restrict__ wgt,
    float* __restrict__ out) {
  const int tid = threadIdx.x;
  const int wav = tid >> 5, lane = tid & 31, lrow = lane & 15, hi = lane >> 4;
  const int b = blockIdx.y;
  const int vt0 = (blockIdx.x * 8 + wav) * 4;

  const __bf16* ar = relT + (size_t)b * 512 + lrow * 32 + hi * 16;
  const v16bf a = cat16(*(const v8bf*)ar, *(const v8bf*)(ar + 8));

#pragma unroll
  for (int t = 0; t < 4; ++t) {
    const int vtile = vt0 + t;
    if (vtile * 16 >= V_) break;              // wave-uniform
    const int v = vtile * 16 + lrow;
    const __bf16* bp = wgt + (size_t)v * 32 + hi * 16;
    const v16bf bb = cat16(*(const v8bf*)bp, *(const v8bf*)(bp + 8));
    v8f acc = {0.f, 0.f, 0.f, 0.f, 0.f, 0.f, 0.f, 0.f};
    acc = __builtin_amdgcn_wmma_f32_16x16x32_bf16(false, a, false, bb,
                                                  (short)0, acc, false, false);
    if (v < V_) {
      float* vp = out + (size_t)(b * V_ + v) * 3;
      const float p0 = vp[0], p1 = vp[1], p2 = vp[2];  // v_posed, read before write
      const float r0 = acc[0] * p0 + acc[1] * p1 + acc[2] * p2 + acc[3];
      const float r1 = acc[4] * p0 + acc[5] * p1 + acc[6] * p2 + acc[7];
      if (hi == 0) { vp[0] = r0; vp[1] = r1; }         // x, y
      else         { vp[2] = r0; }                     // z
    }
  }
}

// ---------------------------------------------------------------------------
extern "C" void kernel_launch(void* const* d_in, const int* in_sizes, int n_in,
                              void* d_out, int out_size, void* d_ws, size_t ws_size,
                              hipStream_t stream) {
  (void)in_sizes; (void)n_in; (void)out_size; (void)ws_size;
  const float* poses    = (const float*)d_in[0];
  const float* shapes   = (const float*)d_in[1];
  const float* vtempl   = (const float*)d_in[2];
  const float* sdirs    = (const float*)d_in[3];
  const float* pdirs    = (const float*)d_in[4];
  const float* jreg     = (const float*)d_in[5];
  const float* weights  = (const float*)d_in[6];
  float* out = (float*)d_out;
  char* ws = (char*)d_ws;

  float*  js   = (float*)(ws + OFF_JS);
  __bf16* X    = (__bf16*)(ws + OFF_X);
  __bf16* relT = (__bf16*)(ws + OFF_RELT);
  __bf16* wgt  = (__bf16*)(ws + OFF_WGT);
  __bf16* wt   = (__bf16*)(ws + OFF_WT);

  k_joint_reg<<<NJ_, 256, 0, stream>>>(jreg, vtempl, sdirs, js);
  k_wpack<<<(VPAD * 32) / 256, 256, 0, stream>>>(weights, wgt);
  k_wtpack<<<NCHP, 256, 0, stream>>>(sdirs, pdirs, wt);
  k_pose<<<B_ / 8, 256, 0, stream>>>(poses, shapes, js, X, relT, out);
  k_gemm<<<dim3(NCHG, B_ / 32), 256, 0, stream>>>(X, wt, vtempl, out);
  k_skin<<<dim3(14, B_), 256, 0, stream>>>(relT, wgt, out);
}
